// _LongcatFlashMLA_85787676770796
// MI455X (gfx1250) — compile-verified, matrix operand order checked by reading, not run on hardware
//
#include <hip/hip_runtime.h>
#include <cstddef>

// ---------------- types ----------------
typedef __attribute__((ext_vector_type(16))) _Float16 v16h;
typedef __attribute__((ext_vector_type(8)))  _Float16 v8h;
typedef __attribute__((ext_vector_type(2)))  _Float16 v2h;
typedef __attribute__((ext_vector_type(8)))  float    v8f;
typedef __attribute__((ext_vector_type(4)))  int      v4i;

// async global->LDS path (CDNA5): use if the toolchain declares the builtins.
// Signature (probe-confirmed via diagnostic): (v4i AS1*, v4i AS3*, imm, imm)
#if __has_builtin(__builtin_amdgcn_global_load_async_to_lds_b128) && \
    __has_builtin(__builtin_amdgcn_s_wait_asynccnt)
#define USE_ASYNC_LDS 1
typedef __attribute__((address_space(1))) v4i gv4i;
typedef __attribute__((address_space(3))) v4i lv4i;
__device__ inline void async_copy16(const _Float16* g, _Float16* l) {
  _Float16* gnc = const_cast<_Float16*>(g);
  __builtin_amdgcn_global_load_async_to_lds_b128(
      (gv4i*)(void*)gnc, (lv4i*)(void*)l, 0, 0);
}
#endif

// ---------------- problem constants ----------------
constexpr int B_SZ    = 2;
constexpr int S_LEN   = 1024;
constexpr int HIDDEN  = 4096;
constexpr int Q_LORA  = 1536;
constexpr int KV_LORA = 512;
constexpr int H_HEADS = 32;
constexpr int QK_NOPE = 128;
constexpr int QK_ROPE = 64;
constexpr int QK_HEAD = QK_NOPE + QK_ROPE;   // 192
constexpr int V_DIM   = 128;
constexpr int TOKENS  = B_SZ * S_LEN;        // 2048

constexpr float EPS        = 1e-6f;
constexpr float Q_SCALE    = 1.63299316185545206547f;  // sqrt(4096/1536)
constexpr float KV_SCALE   = 2.82842712474619009760f;  // sqrt(4096/512)
constexpr float ATTN_SCALE = 0.07216878364870322439f;  // 192^-0.5

// ---------------- WMMA fragment loader ----------------
// CDNA5 16-bit A-operand layout (16x32 MxK): lane L holds row M=L%16.
// Lanes 0-15:  K = {0..7, 16..23}; lanes 16-31: K = {8..15, 24..31}.
// B operand mirrors over N, so loading from an N-major (B^T, N x K) tile
// with the same pattern yields the B fragment.
__device__ inline v16h load_frag(const _Float16* rowBase, int khalf) {
  v8h lo = *reinterpret_cast<const v8h*>(rowBase + khalf);       // k: khalf..+7
  v8h hi = *reinterpret_cast<const v8h*>(rowBase + khalf + 16);  // k: khalf+16..+23
  return __builtin_shufflevector(lo, hi, 0,1,2,3,4,5,6,7,8,9,10,11,12,13,14,15);
}

__device__ inline v8f wmma_f16(v16h a, v16h b, v8f c) {
  return __builtin_amdgcn_wmma_f32_16x16x32_f16(false, a, false, b, (short)0, c,
                                                false, false);
}

__device__ inline v2h pack2(float a, float b) {
  v2h r; r.x = (_Float16)a; r.y = (_Float16)b; return r;
}

// ---------------- weight pre-transpose + f16 convert ----------------
// Bt[n*K + k] = (f16) B[k*N + n].  Tiles: 32(K) x 64(N).
__global__ __launch_bounds__(256)
void transpose_f16(const float* __restrict__ B, _Float16* __restrict__ Bt,
                   int K, int N) {
  __shared__ float tile[32][65];  // pad to dodge bank conflicts
  const int k0 = blockIdx.x * 32, n0 = blockIdx.y * 64;
  for (int idx = threadIdx.x; idx < 32 * 64; idx += 256) {
    const int k = idx >> 6, n = idx & 63;
    tile[k][n] = B[(size_t)(k0 + k) * N + n0 + n];      // coalesced over n
  }
  __syncthreads();
  for (int idx = threadIdx.x; idx < 64 * 16; idx += 256) {
    const int n = idx >> 4, kp = (idx & 15) * 2;
    *reinterpret_cast<v2h*>(&Bt[(size_t)(n0 + n) * K + k0 + kp]) =
        pack2(tile[kp][n], tile[kp + 1][n]);            // packed b32 writes
  }
}

// ---------------- generic WMMA GEMM ----------------
// C = alpha * A(MxK f32, lda) @ B  where B is given pre-transposed f16: Bt (N x K).
// Block tile 128x64, BK=32; 8 waves as 4(M)x2(N), each wave a 32x32 output
// (4 accumulators -> 4 WMMAs per k-step). LDS rows padded to 40 halves.
__global__ __launch_bounds__(256)
void wmma_gemm(const float* __restrict__ A, int lda,
               const _Float16* __restrict__ Bt,
               float* __restrict__ C,
               int M, int N, int K, float alpha) {
  constexpr int BM = 128, BN = 64, BK = 32, LDK = 40;
  __shared__ _Float16 sA[BM * LDK];  // M x K tile, row-major (padded)
  __shared__ _Float16 sB[BN * LDK];  // B^T tile: N x K, row-major (padded)

  const int bn = blockIdx.x, bm = blockIdx.y;
  const int m0 = bm * BM, n0 = bn * BN;
  const int t = threadIdx.x;
  const int wave = t >> 5, lane = t & 31;
  const int wm = wave & 3;        // 0..3 -> M 32-block
  const int wn = wave >> 2;       // 0..1 -> N 32-block
  const int khalf = (lane < 16) ? 0 : 8;

  v8f acc00 = {}, acc01 = {}, acc10 = {}, acc11 = {};

  // staging coords
  const int sa_m = t >> 1, sa_k = (t & 1) * 16;   // A: 16 contiguous floats/thread
  const int sb_n = t >> 2, sb_k = (t & 3) * 8;    // B: 8 contiguous halves/thread

  for (int k0 = 0; k0 < K; k0 += BK) {
    // ---- stage B^T 64(N)x32(K): pure 16B global->LDS copy ----
    {
      const _Float16* gB = Bt + (size_t)(n0 + sb_n) * K + k0 + sb_k;
      _Float16* lB = &sB[sb_n * LDK + sb_k];
#ifdef USE_ASYNC_LDS
      async_copy16(gB, lB);
#else
      *reinterpret_cast<v8h*>(lB) = *reinterpret_cast<const v8h*>(gB);
#endif
    }
    // ---- stage A 128x32: 16 contiguous floats -> packed cvt, two b128 stores ----
    {
      const float* g = A + (size_t)(m0 + sa_m) * lda + k0 + sa_k;
      v8h h0, h1;
      #pragma unroll
      for (int i = 0; i < 8; ++i) h0[i] = (_Float16)g[i];
      #pragma unroll
      for (int i = 0; i < 8; ++i) h1[i] = (_Float16)g[8 + i];
      *reinterpret_cast<v8h*>(&sA[sa_m * LDK + sa_k])     = h0;
      *reinterpret_cast<v8h*>(&sA[sa_m * LDK + sa_k + 8]) = h1;
    }
    // prefetch next K-step tiles into cache (global_prefetch_b8)
    if (k0 + BK < K) {
      __builtin_prefetch(A + (size_t)(m0 + sa_m) * lda + k0 + BK + sa_k, 0, 3);
      __builtin_prefetch(Bt + (size_t)(n0 + sb_n) * K + k0 + BK + sb_k, 0, 3);
    }
#ifdef USE_ASYNC_LDS
    __builtin_amdgcn_s_wait_asynccnt(0);
#endif
    __syncthreads();

    const v16h a0 = load_frag(sA + (wm * 32 +      (lane & 15)) * LDK, khalf);
    const v16h a1 = load_frag(sA + (wm * 32 + 16 + (lane & 15)) * LDK, khalf);
    const v16h b0 = load_frag(sB + (wn * 32 +      (lane & 15)) * LDK, khalf);
    const v16h b1 = load_frag(sB + (wn * 32 + 16 + (lane & 15)) * LDK, khalf);
    acc00 = wmma_f16(a0, b0, acc00);
    acc01 = wmma_f16(a0, b1, acc01);
    acc10 = wmma_f16(a1, b0, acc10);
    acc11 = wmma_f16(a1, b1, acc11);
    __syncthreads();
  }

  // C/D layout: VGPR v -> row (v + 8*(lane>=16)), col = lane&15
  const int n_l   = lane & 15;
  const int mbase = m0 + wm * 32 + ((lane >> 4) << 3);
  const int nb    = n0 + wn * 32 + n_l;
  #pragma unroll
  for (int v = 0; v < 8; ++v) {
    C[(size_t)(mbase + v) * N + nb]           = alpha * acc00[v];
    C[(size_t)(mbase + v) * N + nb + 16]      = alpha * acc01[v];
    C[(size_t)(mbase + 16 + v) * N + nb]      = alpha * acc10[v];
    C[(size_t)(mbase + 16 + v) * N + nb + 16] = alpha * acc11[v];
  }
}

// ---------------- RMSNorm in-place (scale folded) ----------------
__global__ __launch_bounds__(256)
void rmsnorm_inplace(float* __restrict__ x, const float* __restrict__ w,
                     int rowStride, int width, float scale) {
  __shared__ float red[8];
  const int row = blockIdx.x;
  float* p = x + (size_t)row * rowStride;
  float s = 0.f;
  for (int i = threadIdx.x; i < width; i += 256) { float v = p[i]; s += v * v; }
  #pragma unroll
  for (int off = 16; off; off >>= 1) s += __shfl_xor(s, off, 32);
  if ((threadIdx.x & 31) == 0) red[threadIdx.x >> 5] = s;
  __syncthreads();
  float tot = 0.f;
  #pragma unroll
  for (int i = 0; i < 8; ++i) tot += red[i];
  const float inv = rsqrtf(tot / (float)width + EPS) * scale;
  for (int i = threadIdx.x; i < width; i += 256) p[i] = p[i] * inv * w[i];
}

// ---------------- interleaved RoPE ----------------
__global__ void rope_k(float* __restrict__ ckv,
                       const float* __restrict__ cosT, const float* __restrict__ sinT) {
  const int tok = blockIdx.x;
  const int s = tok & (S_LEN - 1);
  const int i = threadIdx.x;  // 0..31 pairs
  float* p = ckv + (size_t)tok * (KV_LORA + QK_ROPE) + KV_LORA;
  const float c = cosT[s * 32 + i], sn = sinT[s * 32 + i];
  const float x1 = p[2 * i], x2 = p[2 * i + 1];
  p[2 * i]     = x1 * c - x2 * sn;
  p[2 * i + 1] = x2 * c + x1 * sn;
}

__global__ __launch_bounds__(256)
void rope_q(float* __restrict__ q,
            const float* __restrict__ cosT, const float* __restrict__ sinT) {
  const int tok = blockIdx.x;
  const int s = tok & (S_LEN - 1);
  for (int idx = threadIdx.x; idx < H_HEADS * 32; idx += 256) {
    const int h = idx >> 5, i = idx & 31;
    float* p = q + (size_t)tok * (H_HEADS * QK_HEAD) + h * QK_HEAD + QK_NOPE;
    const float c = cosT[s * 32 + i], sn = sinT[s * 32 + i];
    const float x1 = p[2 * i], x2 = p[2 * i + 1];
    p[2 * i]     = x1 * c - x2 * sn;
    p[2 * i + 1] = x2 * c + x1 * sn;
  }
}

// ---------------- attention: one block per (b, h, 32-query tile) ----------------
// dynamic LDS layout (bytes), fragment-source tiles use padded row strides:
//   [0,131072)         sS    : 32 x 1024 f32 scores        (stride 1024)
//   [131072,197120)    sP    : 32 x 1032 f16 exp()         (stride 1032, padded)
//   [197120,209920)    sQ    : 32 x 200  f16               (stride 200,  padded)
//   [209920,235520)    sKV   : 64 x 200 f16 K-tiles / 128 x 40 f16 V^T-tiles
//   [235520,235648)    sRinv : 32 f32
constexpr int    SP_STRIDE = 1032;
constexpr int    SQ_STRIDE = 200;
constexpr int    SVT_STRIDE = 40;
constexpr size_t ATTN_LDS_BYTES = 235648;

__global__ __launch_bounds__(256, 1)
void mla_attn(const float* __restrict__ q, const float* __restrict__ kv,
              const float* __restrict__ ckv, const float* __restrict__ bias,
              float* __restrict__ attn) {
  extern __shared__ char smem[];
  float*    sS    = reinterpret_cast<float*>(smem);
  _Float16* sP    = reinterpret_cast<_Float16*>(smem + 131072);
  _Float16* sQ    = reinterpret_cast<_Float16*>(smem + 197120);
  _Float16* sKV   = reinterpret_cast<_Float16*>(smem + 209920);
  float*    sRinv = reinterpret_cast<float*>(smem + 235520);

  const int bid = blockIdx.x;
  const int qb = bid & 31;
  const int h  = (bid >> 5) & 31;
  const int b  = bid >> 10;
  const int tok0 = b * S_LEN + qb * 32;

  const int t = threadIdx.x, wave = t >> 5, lane = t & 31;
  const int khalf = (lane < 16) ? 0 : 8;

  // stage Q tile 32 x 192 as packed pairs
  for (int p = t; p < 32 * QK_HEAD / 2; p += 256) {
    const int idx = p * 2;
    const int m = idx / QK_HEAD, d = idx % QK_HEAD;
    const float* g = q + (size_t)(tok0 + m) * (H_HEADS * QK_HEAD) + h * QK_HEAD + d;
    *reinterpret_cast<v2h*>(&sQ[m * SQ_STRIDE + d]) = pack2(g[0], g[1]);
  }

  // ---- phase 1: scores = Q K^T * scale + bias ----
  {
    const int wm = wave & 1;   // M 16-block (of 32 rows)
    const int wn = wave >> 1;  // N 16-block (of 64 keys)
    for (int kb = 0; kb < S_LEN / 64; ++kb) {
      // stage K tile 64 x 192 as packed pairs (pairs never straddle d=128)
      for (int p = t; p < 64 * QK_HEAD / 2; p += 256) {
        const int idx = p * 2;
        const int j = idx / QK_HEAD, d = idx % QK_HEAD;
        const int key = kb * 64 + j;
        float x0, x1;
        if (d < QK_NOPE) {
          const float* g = kv + (size_t)(b * S_LEN + key) * (H_HEADS * 256) + h * 256 + d;
          x0 = g[0]; x1 = g[1];
        } else {
          const float* g = ckv + (size_t)(b * S_LEN + key) * (KV_LORA + QK_ROPE) +
                           KV_LORA + (d - QK_NOPE);
          x0 = g[0]; x1 = g[1];
        }
        *reinterpret_cast<v2h*>(&sKV[j * SQ_STRIDE + d]) = pack2(x0, x1);
      }
      __syncthreads();

      v8f acc = {};
      const int row_a = wm * 16 + (lane & 15);
      const int row_b = wn * 16 + (lane & 15);
      #pragma unroll
      for (int ks = 0; ks < QK_HEAD / 32; ++ks) {
        const v16h a  = load_frag(sQ  + row_a * SQ_STRIDE + ks * 32, khalf);
        const v16h bb = load_frag(sKV + row_b * SQ_STRIDE + ks * 32, khalf);
        acc = wmma_f16(a, bb, acc);
      }
      const int n_l    = lane & 15;
      const int m_base = wm * 16 + ((lane >> 4) << 3);
      const int kpos   = kb * 64 + wn * 16 + n_l;
      #pragma unroll
      for (int v = 0; v < 8; ++v) {
        const int m = m_base + v;
        const int qpos = qb * 32 + m;
        sS[m * S_LEN + kpos] = acc[v] * ATTN_SCALE +
                               bias[((size_t)b * S_LEN + qpos) * S_LEN + kpos];
      }
      __syncthreads();
    }
  }

  // ---- phase 2: row softmax; store unnormalized exp pairs + 1/sum ----
  {
    #pragma unroll
    for (int r = 0; r < 4; ++r) {
      const int row = wave * 4 + r;
      const float* srow = sS + row * S_LEN;
      float mx = -3.4e38f;
      #pragma unroll
      for (int i = 0; i < S_LEN / 64; ++i) {
        const int c = i * 64 + lane * 2;
        mx = fmaxf(mx, fmaxf(srow[c], srow[c + 1]));
      }
      #pragma unroll
      for (int off = 16; off; off >>= 1) mx = fmaxf(mx, __shfl_xor(mx, off, 32));
      float sum = 0.f;
      #pragma unroll
      for (int i = 0; i < S_LEN / 64; ++i) {
        const int c = i * 64 + lane * 2;
        const float e0 = __expf(srow[c] - mx);
        const float e1 = __expf(srow[c + 1] - mx);
        sum += e0 + e1;
        *reinterpret_cast<v2h*>(&sP[row * SP_STRIDE + c]) = pack2(e0, e1);
      }
      #pragma unroll
      for (int off = 16; off; off >>= 1) sum += __shfl_xor(sum, off, 32);
      if (lane == 0) sRinv[row] = 1.f / sum;
    }
  }
  __syncthreads();

  // ---- phase 3: attn = (P @ V) / rowsum ----
  {
    const int wm = wave & 1;   // M 16-block
    const int wn = wave >> 1;  // 0..3, each 32 of V_DIM=128
    v8f acc0 = {}, acc1 = {};
    for (int kb = 0; kb < S_LEN / 32; ++kb) {
      // stage V^T: 128(dims) x 32(keys), padded stride
      for (int idx = t; idx < V_DIM * 32; idx += 256) {
        const int d = idx & 127, j = idx >> 7;
        const float v = kv[(size_t)(b * S_LEN + kb * 32 + j) * (H_HEADS * 256) +
                           h * 256 + QK_NOPE + d];
        sKV[d * SVT_STRIDE + j] = (_Float16)v;
      }
      __syncthreads();

      const v16h a  = load_frag(sP + (wm * 16 + (lane & 15)) * SP_STRIDE + kb * 32, khalf);
      const v16h b0 = load_frag(sKV + (wn * 32 +      (lane & 15)) * SVT_STRIDE, khalf);
      const v16h b1 = load_frag(sKV + (wn * 32 + 16 + (lane & 15)) * SVT_STRIDE, khalf);
      acc0 = wmma_f16(a, b0, acc0);
      acc1 = wmma_f16(a, b1, acc1);
      __syncthreads();
    }
    const int n_l    = lane & 15;
    const int m_base = wm * 16 + ((lane >> 4) << 3);
    #pragma unroll
    for (int v = 0; v < 8; ++v) {
      const int m = m_base + v;
      const float rinv = sRinv[m];
      const size_t row = (size_t)(tok0 + m) * (H_HEADS * V_DIM) + h * V_DIM;
      attn[row + wn * 32 + n_l]      = acc0[v] * rinv;
      attn[row + wn * 32 + 16 + n_l] = acc1[v] * rinv;
    }
  }
}

// ---------------- host orchestration ----------------
extern "C" void kernel_launch(void* const* d_in, const int* in_sizes, int n_in,
                              void* d_out, int out_size, void* d_ws, size_t ws_size,
                              hipStream_t stream) {
  const float* hidden   = (const float*)d_in[0];   // [2,1024,4096]
  const float* bias     = (const float*)d_in[1];   // [2,1,1024,1024]
  const float* cosT     = (const float*)d_in[2];   // [1024,32]
  const float* sinT     = (const float*)d_in[3];   // [1024,32]
  const float* q_a_w    = (const float*)d_in[4];   // [4096,1536]
  const float* q_a_ln_w = (const float*)d_in[5];   // [1536]
  const float* q_b_w    = (const float*)d_in[6];   // [1536,6144]
  const float* kv_a_w   = (const float*)d_in[7];   // [4096,576]
  const float* kv_a_ln_w= (const float*)d_in[8];   // [512]
  const float* kv_b_w   = (const float*)d_in[9];   // [512,8192]
  const float* o_w      = (const float*)d_in[10];  // [4096,4096]
  float* out = (float*)d_out;                      // [2,1024,4096]

  // workspace carve-up
  float* q_c  = (float*)d_ws;                                  // 2048*1536
  float* ckv  = q_c  + (size_t)TOKENS * Q_LORA;                // 2048*576
  float* qbuf = ckv  + (size_t)TOKENS * (KV_LORA + QK_ROPE);   // 2048*6144
  float* kvb  = qbuf + (size_t)TOKENS * H_HEADS * QK_HEAD;     // 2048*8192
  float* attn = kvb  + (size_t)TOKENS * H_HEADS * 256;         // 2048*4096
  _Float16* q_a_t  = (_Float16*)(attn + (size_t)TOKENS * HIDDEN); // [1536,4096]
  _Float16* kv_a_t = q_a_t  + (size_t)Q_LORA * HIDDEN;            // [576,4096]
  _Float16* q_b_t  = kv_a_t + (size_t)(KV_LORA + QK_ROPE) * HIDDEN;          // [6144,1536]
  _Float16* kv_b_t = q_b_t  + (size_t)(H_HEADS * QK_HEAD) * Q_LORA;          // [8192,512]
  _Float16* o_t    = kv_b_t + (size_t)(H_HEADS * 256) * KV_LORA;             // [4096,4096]

  (void)in_sizes; (void)n_in; (void)out_size; (void)ws_size;

  (void)hipFuncSetAttribute(reinterpret_cast<const void*>(mla_attn),
                            hipFuncAttributeMaxDynamicSharedMemorySize,
                            (int)ATTN_LDS_BYTES);

  const dim3 blk(256);

  // 0) pre-transpose + f16-convert weights (amortized over all GEMM blocks)
  transpose_f16<<<dim3(HIDDEN / 32,  Q_LORA / 64),              blk, 0, stream>>>(q_a_w,  q_a_t,  HIDDEN,  Q_LORA);
  transpose_f16<<<dim3(HIDDEN / 32,  (KV_LORA + QK_ROPE) / 64), blk, 0, stream>>>(kv_a_w, kv_a_t, HIDDEN,  KV_LORA + QK_ROPE);
  transpose_f16<<<dim3(Q_LORA / 32,  H_HEADS * QK_HEAD / 64),   blk, 0, stream>>>(q_b_w,  q_b_t,  Q_LORA,  H_HEADS * QK_HEAD);
  transpose_f16<<<dim3(KV_LORA / 32, H_HEADS * 256 / 64),       blk, 0, stream>>>(kv_b_w, kv_b_t, KV_LORA, H_HEADS * 256);
  transpose_f16<<<dim3(HIDDEN / 32,  HIDDEN / 64),              blk, 0, stream>>>(o_w,    o_t,    HIDDEN,  HIDDEN);

  // 1) q_c_pre = hidden @ q_a_w
  wmma_gemm<<<dim3(Q_LORA / 64, TOKENS / 128), blk, 0, stream>>>(
      hidden, HIDDEN, q_a_t, q_c, TOKENS, Q_LORA, HIDDEN, 1.0f);
  // 2) ckv = hidden @ kv_a_w
  wmma_gemm<<<dim3((KV_LORA + QK_ROPE) / 64, TOKENS / 128), blk, 0, stream>>>(
      hidden, HIDDEN, kv_a_t, ckv, TOKENS, KV_LORA + QK_ROPE, HIDDEN, 1.0f);
  // 3) rmsnorm q_c (in place)
  rmsnorm_inplace<<<TOKENS, blk, 0, stream>>>(q_c, q_a_ln_w, Q_LORA, Q_LORA, 1.0f);
  // 4) rmsnorm k_pass = ckv[:, :512] (in place, * KV_SCALE)
  rmsnorm_inplace<<<TOKENS, blk, 0, stream>>>(ckv, kv_a_ln_w, KV_LORA + QK_ROPE,
                                              KV_LORA, KV_SCALE);
  // 5) rope on k_rope = ckv[:, 512:576] (in place)
  rope_k<<<TOKENS, dim3(32), 0, stream>>>(ckv, cosT, sinT);
  // 6) q = (q_c @ q_b_w) * Q_SCALE  (scalar commutes with RoPE)
  wmma_gemm<<<dim3(H_HEADS * QK_HEAD / 64, TOKENS / 128), blk, 0, stream>>>(
      q_c, Q_LORA, q_b_t, qbuf, TOKENS, H_HEADS * QK_HEAD, Q_LORA, Q_SCALE);
  // 7) rope on q_rope (in place)
  rope_q<<<TOKENS, blk, 0, stream>>>(qbuf, cosT, sinT);
  // 8) kv = k_pass @ kv_b_w   (A row stride 576, width 512)
  wmma_gemm<<<dim3(H_HEADS * 256 / 64, TOKENS / 128), blk, 0, stream>>>(
      ckv, KV_LORA + QK_ROPE, kv_b_t, kvb, TOKENS, H_HEADS * 256, KV_LORA, 1.0f);
  // 9) attention
  mla_attn<<<dim3(B_SZ * H_HEADS * (S_LEN / 32)), blk, ATTN_LDS_BYTES, stream>>>(
      qbuf, kvb, ckv, bias, attn);
  // 10) out = attn @ o_w
  wmma_gemm<<<dim3(HIDDEN / 64, TOKENS / 128), blk, 0, stream>>>(
      attn, H_HEADS * V_DIM, o_t, out, TOKENS, HIDDEN, HIDDEN, 1.0f);
}
